// MyLSTM_4672924418479
// MI455X (gfx1250) — matrix operand order repaired and checked by reading, not compiled, not run on hardware
//
#include <hip/hip_runtime.h>
#include <hip/hip_bf16.h>

// ---------------------------------------------------------------------------
// Persistent WMMA-bf16 LSTM for MI455X (gfx1250, wave32).
//   B=64, T=512, ID=512, HD=1024.  gates[b,g] = sum_k A[b,k]*Wcat[g,k] + bias
//   with A = [q_{t-1} (1024) | x_t (512)],  Wcat = [W_hh | W_ih].
// 256 persistent blocks x 128 threads (4 waves). Each block owns 4 hidden
// columns (4 gates x 4 cols = one 16-wide N tile), weights resident in LDS,
// cell state in registers, device-wide spin barrier per timestep.
// ---------------------------------------------------------------------------

#define B_    64
#define T_    512
#define ID_   512
#define HD_   1024
#define KQ_   1024          // K range covered by q / W_hh
#define KKQ   32            // k-steps (of 32) in q part
#define KKT   48            // total k-steps (q:32 + x:16)
#define NBLK  256           // persistent blocks
#define NW    4             // hidden columns per block  (NBLK*NW == HD_)
#define THREADS 128         // 4 waves of 32
#define EPB   (KKT * 2 * 32 * 8)   // packed weight elems per block = 24576

typedef __attribute__((ext_vector_type(16))) __bf16 v16bf;
typedef __attribute__((ext_vector_type(8)))  float  v8f;

union Frag { v16bf v; uint4 u[2]; };

__device__ __forceinline__ unsigned short f2bf(float f) {
  unsigned u = __float_as_uint(f);
  u += 0x7FFFu + ((u >> 16) & 1u);       // round-to-nearest-even
  return (unsigned short)(u >> 16);
}
// Branch-free gate nonlinearities (v_exp_f32 + v_rcp_f32, no libm divergence).
__device__ __forceinline__ float sigm(float x) {
  return __builtin_amdgcn_rcpf(1.0f + __expf(-x));
}
__device__ __forceinline__ float tanh_fast(float x) {
  // tanh(x) = 1 - 2/(1 + exp(2x)); exact at +/-inf, monotone, ~1ulp of rcp.
  return 1.0f - 2.0f * __builtin_amdgcn_rcpf(1.0f + __expf(2.0f * x));
}

// --------------------------- prep kernels ----------------------------------

__global__ void init_state_kernel(unsigned int* cnt, unsigned short* qbuf) {
  int i = blockIdx.x * blockDim.x + threadIdx.x;
  if (i == 0) *cnt = 0u;
  const int n = 2 * B_ * HD_;
  for (int j = i; j < n; j += gridDim.x * blockDim.x) qbuf[j] = 0;  // q_{-1}=0
}

__global__ void pack_x_kernel(const float* __restrict__ x0,
                              unsigned short* __restrict__ x0bf) {
  const int n = B_ * T_ * ID_;
  for (int i = blockIdx.x * blockDim.x + threadIdx.x; i < n;
       i += gridDim.x * blockDim.x)
    x0bf[i] = f2bf(x0[i]);
}

// Pack Wcat into per-block B-operand fragments in this kernel's contiguous-k B order:
// flat idx = ((blk*KKT + kk)*2 + c)*32*8 + L*8 + j
//   lane L: column p = L%16, half = L/16 ;  element e = c*8+j -> k = kk*32 + 16*half + e
//   packed column p: gate = p/4 (0=f,1=i,2=a,3=o), h-col = blk*NW + p%4
__global__ void pack_w_kernel(const float* __restrict__ W_ih,
                              const float* __restrict__ W_hh,
                              unsigned short* __restrict__ wpack) {
  const int n = NBLK * EPB;
  for (int idx = blockIdx.x * blockDim.x + threadIdx.x; idx < n;
       idx += gridDim.x * blockDim.x) {
    int j    = idx & 7;
    int L    = (idx >> 3) & 31;
    int c    = (idx >> 8) & 1;
    int kk   = (idx >> 9) % KKT;
    int blk  = idx / EPB;
    int half = L >> 4;
    int p    = L & 15;
    int g    = (p >> 2) * HD_ + blk * NW + (p & 3);
    int k    = kk * 32 + 16 * half + c * 8 + j;
    float v  = (k < KQ_) ? W_hh[(size_t)g * HD_ + k]
                         : W_ih[(size_t)g * ID_ + (k - KQ_)];
    wpack[idx] = f2bf(v);
  }
}

__global__ void pack_bias_kernel(const float* __restrict__ b_ih,
                                 const float* __restrict__ b_hh,
                                 float* __restrict__ bpack) {
  int i = blockIdx.x * blockDim.x + threadIdx.x;   // NBLK*16
  if (i >= NBLK * 16) return;
  int blk = i >> 4, p = i & 15;
  int g = (p >> 2) * HD_ + blk * NW + (p & 3);
  bpack[i] = b_ih[g] + b_hh[g];
}

// --------------------------- persistent LSTM -------------------------------

__global__ __launch_bounds__(THREADS) void
lstm_persist_kernel(const unsigned short* __restrict__ x0bf,
                    const unsigned short* __restrict__ wpack,
                    const float* __restrict__ bpack,
                    unsigned short* __restrict__ qbuf,   // [2][B][HD] bf16
                    float* __restrict__ out,
                    unsigned int* __restrict__ cnt) {
  __shared__ __align__(16) unsigned short wlds[EPB];     // 48 KB weights

  const int blk  = blockIdx.x;
  const int tid  = threadIdx.x;
  const int wave = tid >> 5;          // m-tile (batch rows wave*16..+15)
  const int lane = tid & 31;
  const int nloc = lane & 15;         // packed column within N tile
  const int half = lane >> 4;

  // One-time: stage this block's weight slice into LDS (straight copy).
  {
    const uint4* src = (const uint4*)wpack + (size_t)blk * (EPB / 8);
    uint4* dst = (uint4*)wlds;
    for (int i = tid; i < EPB / 8; i += THREADS) dst[i] = src[i];
  }
  __syncthreads();

  const float bias = bpack[blk * 16 + nloc];

  float sreg[8];
#pragma unroll
  for (int v = 0; v < 8; ++v) sreg[v] = 0.0f;

  const size_t QLAST = (size_t)B_ * T_ * HD_;
  const size_t SLAST = QLAST + (size_t)B_ * HD_;

  const int arow = wave * 16 + nloc;  // A-operand row (batch) this lane loads
  const uint4* ldsv = (const uint4*)wlds;

  for (int t = 0; t < T_; ++t) {
    const unsigned short* qsrc  = qbuf + (size_t)((t + 1) & 1) * (B_ * HD_);
    const unsigned short* arow_q = qsrc + (size_t)arow * HD_;
    const unsigned short* arow_x = x0bf + ((size_t)arow * T_ + t) * ID_;

    v8f acc = {};
    for (int kk = 0; kk < KKT; ++kk) {
      const unsigned short* ap;
      int kb;
      if (kk < KKQ) { ap = arow_q; kb = kk * 32; }
      else          { ap = arow_x; kb = (kk - KKQ) * 32; }
      Frag A, Bf;
      // A layout: e -> k = kb + 16*half + e, contiguous per lane
      A.u[0] = *(const uint4*)(ap + kb + 16 * half);
      A.u[1] = *(const uint4*)(ap + kb + 16 * half + 8);
      // B layout: e -> k = kb + 16*half + e (pre-packed), contiguous per lane
      Bf.u[0] = ldsv[(kk * 2 + 0) * 32 + lane];
      Bf.u[1] = ldsv[(kk * 2 + 1) * 32 + lane];
      acc = __builtin_amdgcn_wmma_f32_16x16x32_bf16(
          false, A.v, false, Bf.v, (short)0, acc, false, false);
    }

#pragma unroll
    for (int v = 0; v < 8; ++v) acc[v] += bias;

    // D layout: row m = v + 8*half (within tile), col = nloc.
    // cols 0-3:f, 4-7:i, 8-11:a, 12-15:o  -> lanes nloc<4 own a hidden column.
    unsigned short* qdst = qbuf + (size_t)(t & 1) * (B_ * HD_);
    const int hcol = blk * NW + nloc;
#pragma unroll
    for (int v = 0; v < 8; ++v) {
      float g_f = acc[v];
      float g_i = __shfl_xor(acc[v], 4, 32);
      float g_a = __shfl_xor(acc[v], 8, 32);
      float g_o = __shfl_xor(acc[v], 12, 32);
      if (nloc < 4) {
        float f  = sigm(g_f);
        float ii = sigm(g_i);
        float a  = tanh_fast(g_a);
        float o  = sigm(g_o);
        float sn = f * sreg[v] + ii * a;
        float qn = o * tanh_fast(sn);
        sreg[v] = sn;
        int b = wave * 16 + v + 8 * half;
        out[((size_t)b * T_ + t) * HD_ + hcol] = qn;
        qdst[(size_t)b * HD_ + hcol] = f2bf(qn);
        if (t == T_ - 1) {
          out[QLAST + (size_t)b * HD_ + hcol] = qn;
          out[SLAST + (size_t)b * HD_ + hcol] = sn;
        }
      }
    }

    // Device-wide barrier: all blocks must publish q_t before step t+1.
    __threadfence();
    __syncthreads();
    if (tid == 0) {
      __hip_atomic_fetch_add(cnt, 1u, __ATOMIC_RELEASE, __HIP_MEMORY_SCOPE_AGENT);
      const unsigned tgt = (unsigned)NBLK * (unsigned)(t + 1);
      while (__hip_atomic_load(cnt, __ATOMIC_ACQUIRE, __HIP_MEMORY_SCOPE_AGENT) < tgt)
        __builtin_amdgcn_s_sleep(2);
    }
    __syncthreads();
    __threadfence();   // acquire-side invalidate before reading q_t
  }
}

// ------------------------------ launch -------------------------------------

extern "C" void kernel_launch(void* const* d_in, const int* in_sizes, int n_in,
                              void* d_out, int out_size, void* d_ws, size_t ws_size,
                              hipStream_t stream) {
  (void)in_sizes; (void)n_in; (void)out_size; (void)ws_size;
  const float* x0   = (const float*)d_in[0];
  const float* W_ih = (const float*)d_in[1];
  const float* b_ih = (const float*)d_in[2];
  const float* W_hh = (const float*)d_in[3];
  const float* b_hh = (const float*)d_in[4];
  float* out = (float*)d_out;

  char* ws = (char*)d_ws;
  unsigned int*   cnt   = (unsigned int*)ws;                         // 256 B
  unsigned short* qbuf  = (unsigned short*)(ws + 256);               // 256 KB
  float*          bpack = (float*)(ws + 256 + 2 * B_ * HD_ * 2);     // 16 KB
  unsigned short* x0bf  = (unsigned short*)((char*)bpack + NBLK * 16 * 4); // 32 MB
  unsigned short* wpack = x0bf + (size_t)B_ * T_ * ID_;              // 12.6 MB

  init_state_kernel<<<64, 256, 0, stream>>>(cnt, qbuf);
  pack_x_kernel<<<4096, 256, 0, stream>>>(x0, x0bf);
  pack_w_kernel<<<4096, 256, 0, stream>>>(W_ih, W_hh, wpack);
  pack_bias_kernel<<<16, 256, 0, stream>>>(b_ih, b_hh, bpack);
  lstm_persist_kernel<<<NBLK, THREADS, 0, stream>>>(x0bf, wpack, bpack,
                                                    qbuf, out, cnt);
}
